// RelPositionMultiHeadAttention_81097572483665
// MI455X (gfx1250) — matrix operand order repaired and checked by reading, not compiled, not run on hardware
//
#include <hip/hip_runtime.h>
#include <hip/hip_bf16.h>

// CDNA5 / gfx1250 wave32 WMMA implementation of RelPositionMultiHeadAttention.
// Pipeline: f32->bf16 conversion pass, then all GEMMs as v_wmma_f32_16x16x32_bf16
// (f32 accumulate), flash-style attention with analytic rel_shift band GEMM.

typedef __attribute__((ext_vector_type(16))) __bf16 v16bf;
typedef __attribute__((ext_vector_type(8)))  float  v8f;

union Frag {
    v16bf v;
    __bf16 e[16];
    uint4 q[2];
};

__device__ __forceinline__ unsigned short f2bf(float f) {
    union { float f; unsigned u; } x; x.f = f;
    unsigned r = x.u + 0x7FFFu + ((x.u >> 16) & 1u);   // round-to-nearest-even
    return (unsigned short)(r >> 16);
}

#define NF 512

// ---------------------------------------------------------------------------
// Bulk f32 -> bf16 conversion (8 elements / thread, b128 in / b128 out).
// ---------------------------------------------------------------------------
__global__ __launch_bounds__(256)
void cvt_bf16(const float* __restrict__ in, unsigned short* __restrict__ out, int n)
{
    const int i = (blockIdx.x * 256 + threadIdx.x) * 8;
    if (i + 8 > n) return;
    float4 a = *(const float4*)(in + i);
    float4 b = *(const float4*)(in + i + 4);
    union { unsigned short s[8]; uint4 q; } o;
    o.s[0] = f2bf(a.x); o.s[1] = f2bf(a.y); o.s[2] = f2bf(a.z); o.s[3] = f2bf(a.w);
    o.s[4] = f2bf(b.x); o.s[5] = f2bf(b.y); o.s[6] = f2bf(b.z); o.s[7] = f2bf(b.w);
    *(uint4*)(out + i) = o.q;
}

// ---------------------------------------------------------------------------
// GEMM wave kernel: C(16x64 per wave) = A(Mx512,bf16) * W(512x512,bf16)^T + bias
// One wave per block; fragments loaded straight from global (L2-resident operands),
// no LDS, no barriers. MODE is a compile-time specialization:
//   0: out0 bf16 [b,h,t,d]          (k projection)
//   1: out0/out1 bf16 [b,h,t,d] = val + pos_bias_u / pos_bias_v (q projection)
//   2: out0 bf16 [h, l(=2047), d]   (pos projection, no bias)
//   3: out0 f32 row-major [m, n]    (final projection)
//   4: out0 bf16 [b,h,d,t]          (v projection, transposed for the PV WMMA)
// ---------------------------------------------------------------------------
template<int MODE>
__global__ __launch_bounds__(32)
void proj_gemm(const unsigned short* __restrict__ A,
               const unsigned short* __restrict__ Wb,
               const float* __restrict__ bias,
               const float* __restrict__ pbu, const float* __restrict__ pbv,
               void* __restrict__ out0, void* __restrict__ out1,
               int M)
{
    const int lane = threadIdx.x & 31;
    const int l15  = lane & 15;
    const int half = lane >> 4;
    const int hi8  = half * 8;
    const int m0   = blockIdx.x * 16;
    const int n0   = blockIdx.y * 64;

    const int  arow = m0 + l15;
    const bool aok  = arow < M;
    const int  ar   = aok ? arow : (M - 1);

    v8f acc[4] = {};

    for (int k0 = 0; k0 < NF; k0 += 32) {
        // A fragment: 16x32 bf16, lane = row (l15), K split by half-wave.
        Frag a;
        const unsigned short* ap = A + (size_t)ar * NF + k0 + half*16;
        a.q[0] = *(const uint4*)(ap);        // K: k0+16*half   .. +7
        a.q[1] = *(const uint4*)(ap + 8);    // K: k0+16*half+8 .. +15
        if (!aok) { a.q[0] = make_uint4(0,0,0,0); a.q[1] = make_uint4(0,0,0,0); }

        #pragma unroll
        for (int t = 0; t < 4; ++t) {
            // B fragment: W^T tile (32K x 16N). lane = output col, K contiguous.
            Frag bb;
            const unsigned short* wp = Wb + (size_t)(n0 + t*16 + l15) * NF + k0 + half*16;
            bb.q[0] = *(const uint4*)(wp);
            bb.q[1] = *(const uint4*)(wp + 8);
            acc[t] = __builtin_amdgcn_wmma_f32_16x16x32_bf16(
                false, a.v, false, bb.v, (short)0, acc[t], false, false);
        }
    }

    // Epilogue: C layout -> lane = col (l15), rows r + 8*half.
    #pragma unroll
    for (int t = 0; t < 4; ++t) {
        const int ncol = n0 + t*16 + l15;
        const float bvv = (MODE == 2) ? 0.0f : bias[ncol];
        float pu = 0.0f, pv = 0.0f;
        if (MODE == 1) { pu = pbu[ncol]; pv = pbv[ncol]; }
        const int hh = ncol >> 6, dd = ncol & 63;
        #pragma unroll
        for (int r = 0; r < 8; ++r) {
            const int mrow = m0 + r + hi8;
            if (mrow >= M) continue;
            const float val = acc[t][r] + bvv;
            if (MODE == 3) {
                ((float*)out0)[(size_t)mrow * NF + ncol] = val;
            } else if (MODE == 2) {
                ((unsigned short*)out0)[((size_t)hh * 2047 + mrow) * 64 + dd] = f2bf(val);
            } else {
                const int bidx = mrow >> 10, tt = mrow & 1023;
                if (MODE == 4) {
                    ((unsigned short*)out0)[((size_t)(bidx*8 + hh)*64 + dd) * 1024 + tt] = f2bf(val);
                } else if (MODE == 0) {
                    ((unsigned short*)out0)[((size_t)(bidx*8 + hh)*1024 + tt) * 64 + dd] = f2bf(val);
                } else { // MODE 1: q_u and q_v
                    const size_t adr = ((size_t)(bidx*8 + hh)*1024 + tt) * 64 + dd;
                    ((unsigned short*)out0)[adr] = f2bf(val + pu);
                    ((unsigned short*)out1)[adr] = f2bf(val + pv);
                }
            }
        }
    }
}

// ---------------------------------------------------------------------------
// Flash attention with relative-position band term.
// One wave handles 16 q-rows of one (b,h). Streams 64-key tiles.
// scores[q,k] = (qu.K[k] + qv.P[T-1+k-q]) / 8 ; masked -> -10000 ; online softmax ; O += P.V
// rel_shift is computed analytically via the banded P-GEMM (16x96 window).
// ---------------------------------------------------------------------------
__global__ __launch_bounds__(32)
void attn_kernel(const unsigned short* __restrict__ QU,
                 const unsigned short* __restrict__ QV,
                 const unsigned short* __restrict__ KK,
                 const unsigned short* __restrict__ VT,
                 const unsigned short* __restrict__ PP,
                 const unsigned char*  __restrict__ MSK,
                 unsigned short* __restrict__ XO)
{
    __shared__ float          bdscr[16][96];   // banded bd result (wave-private)
    __shared__ unsigned short pscr[16][72];    // probs, C-layout -> A-layout bounce

    const int lane = threadIdx.x & 31;
    const int l15  = lane & 15;
    const int half = lane >> 4;
    const int hi8  = half * 8;
    const int q0   = blockIdx.x * 16;
    const int bh   = blockIdx.y;
    const int b    = bh >> 3, h = bh & 7;

    const unsigned short* quB = QU + (size_t)bh * 1024 * 64;
    const unsigned short* qvB = QV + (size_t)bh * 1024 * 64;
    const unsigned short* kB  = KK + (size_t)bh * 1024 * 64;
    const unsigned short* vB  = VT + (size_t)bh * 64 * 1024;   // [d][t]
    const unsigned short* pB  = PP + (size_t)h * 2047 * 64;

    // Preload Qu / Qv A-fragments (16x64 each -> 2 frags).
    Frag aQu[2], aQv[2];
    #pragma unroll
    for (int f = 0; f < 2; ++f) {
        const unsigned short* p1 = quB + (size_t)(q0 + l15)*64 + f*32 + half*16;
        aQu[f].q[0] = *(const uint4*)(p1);
        aQu[f].q[1] = *(const uint4*)(p1 + 8);
        const unsigned short* p2 = qvB + (size_t)(q0 + l15)*64 + f*32 + half*16;
        aQv[f].q[0] = *(const uint4*)(p2);
        aQv[f].q[1] = *(const uint4*)(p2 + 8);
    }

    float mrun[8], lrun[8];
    v8f O[4] = {};
    #pragma unroll
    for (int r = 0; r < 8; ++r) { mrun[r] = -3.0e38f; lrun[r] = 0.0f; }

    for (int k0 = 0; k0 < 1024; k0 += 64) {
        // ---- AC = Qu * K_tile^T (16x64), 8 WMMAs ----
        v8f s[4] = {};
        #pragma unroll
        for (int f = 0; f < 2; ++f) {
            #pragma unroll
            for (int t = 0; t < 4; ++t) {
                Frag bk;
                const unsigned short* p = kB + (size_t)(k0 + t*16 + l15)*64 + f*32 + half*16;
                bk.q[0] = *(const uint4*)(p);
                bk.q[1] = *(const uint4*)(p + 8);
                s[t] = __builtin_amdgcn_wmma_f32_16x16x32_bf16(
                    false, aQu[f].v, false, bk.v, (short)0, s[t], false, false);
            }
        }

        // ---- BD = Qv * P_band^T (16x96 window), 12 WMMAs ----
        {
            const int l0 = 1023 + k0 - q0 - 15;   // band start (>= 0 here)
            v8f bd[6] = {};
            #pragma unroll
            for (int f = 0; f < 2; ++f) {
                #pragma unroll
                for (int ct = 0; ct < 6; ++ct) {
                    int l = l0 + ct*16 + l15;
                    l = (l > 2046) ? 2046 : l;    // padded cols never extracted
                    Frag bp;
                    const unsigned short* p = pB + (size_t)l*64 + f*32 + half*16;
                    bp.q[0] = *(const uint4*)(p);
                    bp.q[1] = *(const uint4*)(p + 8);
                    bd[ct] = __builtin_amdgcn_wmma_f32_16x16x32_bf16(
                        false, aQv[f].v, false, bp.v, (short)0, bd[ct], false, false);
                }
            }
            #pragma unroll
            for (int ct = 0; ct < 6; ++ct)
                #pragma unroll
                for (int r = 0; r < 8; ++r)
                    bdscr[r + hi8][ct*16 + l15] = bd[ct][r];
        }

        // ---- combine + diagonal extract + mask (bd[q,k] at window col kc-qi+15) ----
        const unsigned char* mrow = MSK + (size_t)b*1024*1024 + (size_t)q0*1024 + k0;
        #pragma unroll
        for (int t = 0; t < 4; ++t) {
            #pragma unroll
            for (int r = 0; r < 8; ++r) {
                const int qi = r + hi8;
                const int kc = t*16 + l15;
                const float sc = (s[t][r] + bdscr[qi][kc - qi + 15]) * 0.125f;
                const unsigned char mb = mrow[(size_t)qi*1024 + kc];
                s[t][r] = mb ? -10000.0f : sc;   // exp(-1e4 - m) == 0.f in fp32
            }
        }

        // ---- online softmax: row stats via shfl within 16-lane halves ----
        float tmax[8];
        #pragma unroll
        for (int r = 0; r < 8; ++r)
            tmax[r] = fmaxf(fmaxf(s[0][r], s[1][r]), fmaxf(s[2][r], s[3][r]));
        #pragma unroll
        for (int sh = 1; sh <= 8; sh <<= 1)
            #pragma unroll
            for (int r = 0; r < 8; ++r)
                tmax[r] = fmaxf(tmax[r], __shfl_xor(tmax[r], sh, 32));

        float alpha[8], rsum[8];
        #pragma unroll
        for (int r = 0; r < 8; ++r) {
            const float mn = fmaxf(mrun[r], tmax[r]);
            alpha[r] = __expf(mrun[r] - mn);
            mrun[r]  = mn;
            rsum[r]  = 0.0f;
        }
        #pragma unroll
        for (int t = 0; t < 4; ++t)
            #pragma unroll
            for (int r = 0; r < 8; ++r) {
                const float p = __expf(s[t][r] - mrun[r]);
                s[t][r] = p;
                rsum[r] += p;
            }
        #pragma unroll
        for (int sh = 1; sh <= 8; sh <<= 1)
            #pragma unroll
            for (int r = 0; r < 8; ++r)
                rsum[r] += __shfl_xor(rsum[r], sh, 32);
        #pragma unroll
        for (int r = 0; r < 8; ++r) lrun[r] = lrun[r]*alpha[r] + rsum[r];
        #pragma unroll
        for (int t = 0; t < 4; ++t)
            #pragma unroll
            for (int r = 0; r < 8; ++r)
                O[t][r] *= alpha[r];

        // ---- probs C-layout -> bf16 A-fragments via LDS (same-wave, in-order DS) ----
        #pragma unroll
        for (int t = 0; t < 4; ++t)
            #pragma unroll
            for (int r = 0; r < 8; ++r)
                pscr[r + hi8][t*16 + l15] = f2bf(s[t][r]);

        Frag aP[2];
        #pragma unroll
        for (int f = 0; f < 2; ++f) {
            aP[f].q[0] = *(const uint4*)(&pscr[l15][f*32 + half*16]);
            aP[f].q[1] = *(const uint4*)(&pscr[l15][f*32 + half*16 + 8]);
        }

        // ---- O += P * V (V stored transposed [d][t] -> contiguous B frags), 8 WMMAs ----
        #pragma unroll
        for (int f = 0; f < 2; ++f) {
            #pragma unroll
            for (int t = 0; t < 4; ++t) {
                Frag bvf;
                const unsigned short* p = vB + (size_t)(t*16 + l15)*1024 + k0 + f*32 + half*16;
                bvf.q[0] = *(const uint4*)(p);
                bvf.q[1] = *(const uint4*)(p + 8);
                O[t] = __builtin_amdgcn_wmma_f32_16x16x32_bf16(
                    false, aP[f].v, false, bvf.v, (short)0, O[t], false, false);
            }
        }
    }

    // ---- normalize and write X in [b, t, h*64+d] bf16 for the final projection ----
    #pragma unroll
    for (int t = 0; t < 4; ++t)
        #pragma unroll
        for (int r = 0; r < 8; ++r) {
            const int qi = r + hi8;
            const float inv = (lrun[r] != 0.0f) ? 1.0f / lrun[r] : 0.0f;
            XO[((size_t)(b*1024 + q0 + qi))*512 + h*64 + t*16 + l15] = f2bf(O[t][r] * inv);
        }
}

// ---------------------------------------------------------------------------
extern "C" void kernel_launch(void* const* d_in, const int* in_sizes, int n_in,
                              void* d_out, int out_size, void* d_ws, size_t ws_size,
                              hipStream_t stream)
{
    const float* query   = (const float*)d_in[0];
    const float* key     = (const float*)d_in[1];
    const float* value   = (const float*)d_in[2];
    const unsigned char* mask = (const unsigned char*)d_in[3];  // jnp bool = 1 byte
    const float* pos_emb = (const float*)d_in[4];
    const float* Wq  = (const float*)d_in[5];
    const float* bq  = (const float*)d_in[6];
    const float* Wk  = (const float*)d_in[7];
    const float* bk  = (const float*)d_in[8];
    const float* Wv  = (const float*)d_in[9];
    const float* bv  = (const float*)d_in[10];
    const float* Wpos= (const float*)d_in[11];
    const float* Wo  = (const float*)d_in[12];
    const float* bo  = (const float*)d_in[13];
    const float* pbu = (const float*)d_in[14];
    const float* pbv = (const float*)d_in[15];

    // Workspace layout (bytes). Dead regions are reused by later kernels
    // (stream-ordered, so writes never race the kernels that read them):
    //   [ 0, 8M)  qbf  -> reused as kb  (k proj output) after proj_q
    //   [ 8,16M)  kbf  -> reused as xb  (attention output) after proj_k
    //   [16,24M)  vbf
    //   [24,26M)  posbf
    //   [26,29M)  5 bf16 weight matrices (512 KiB each)
    //   [29,37M)  qu   [37,45M) qv   [45,53M) vt   [53,55M) pb
    char* ws = (char*)d_ws;
    unsigned short* qbf   = (unsigned short*)(ws + ((size_t)0  << 20));
    unsigned short* kbf   = (unsigned short*)(ws + ((size_t)8  << 20));
    unsigned short* vbf   = (unsigned short*)(ws + ((size_t)16 << 20));
    unsigned short* posbf = (unsigned short*)(ws + ((size_t)24 << 20));
    unsigned short* wq    = (unsigned short*)(ws + ((size_t)26 << 20));
    unsigned short* wk    = wq   + (512*512);
    unsigned short* wv    = wk   + (512*512);
    unsigned short* wpos  = wv   + (512*512);
    unsigned short* wo    = wpos + (512*512);
    unsigned short* qu    = (unsigned short*)(ws + ((size_t)29 << 20));
    unsigned short* qv    = (unsigned short*)(ws + ((size_t)37 << 20));
    unsigned short* vt    = (unsigned short*)(ws + ((size_t)45 << 20));
    unsigned short* pb    = (unsigned short*)(ws + ((size_t)53 << 20));
    unsigned short* kb    = qbf;   // reuse
    unsigned short* xb    = kbf;   // reuse

    const dim3 blk(32);
    const int NBT = 8*1024*512;        // 4.19M elems
    const int NPOS = 2047*512;
    const int NW  = 512*512;

    // f32 -> bf16 conversion passes (bandwidth-trivial)
    cvt_bf16<<<dim3((NBT  + 2047)/2048), dim3(256), 0, stream>>>(query,   qbf,   NBT);
    cvt_bf16<<<dim3((NBT  + 2047)/2048), dim3(256), 0, stream>>>(key,     kbf,   NBT);
    cvt_bf16<<<dim3((NBT  + 2047)/2048), dim3(256), 0, stream>>>(value,   vbf,   NBT);
    cvt_bf16<<<dim3((NPOS + 2047)/2048), dim3(256), 0, stream>>>(pos_emb, posbf, NPOS);
    cvt_bf16<<<dim3((NW   + 2047)/2048), dim3(256), 0, stream>>>(Wq,   wq,   NW);
    cvt_bf16<<<dim3((NW   + 2047)/2048), dim3(256), 0, stream>>>(Wk,   wk,   NW);
    cvt_bf16<<<dim3((NW   + 2047)/2048), dim3(256), 0, stream>>>(Wv,   wv,   NW);
    cvt_bf16<<<dim3((NW   + 2047)/2048), dim3(256), 0, stream>>>(Wpos, wpos, NW);
    cvt_bf16<<<dim3((NW   + 2047)/2048), dim3(256), 0, stream>>>(Wo,   wo,   NW);

    // q_u / q_v = query*Wq^T + bq (+pos_bias_u / +pos_bias_v), head-major bf16
    proj_gemm<1><<<dim3(512, 8), blk, 0, stream>>>(qbf, wq, bq, pbu, pbv, qu, qv, 8192);
    // k  [b,h,t,d]
    proj_gemm<0><<<dim3(512, 8), blk, 0, stream>>>(kbf, wk, bk, nullptr, nullptr, kb, nullptr, 8192);
    // v  transposed [b,h,d,t]
    proj_gemm<4><<<dim3(512, 8), blk, 0, stream>>>(vbf, wv, bv, nullptr, nullptr, vt, nullptr, 8192);
    // p = pos_emb*Wpos^T, [h, 2047, d]
    proj_gemm<2><<<dim3(128, 8), blk, 0, stream>>>(posbf, wpos, nullptr, nullptr, nullptr, pb, nullptr, 2047);
    // flash attention: 64 q-tiles x 64 (b,h)
    attn_kernel<<<dim3(64, 64), blk, 0, stream>>>(qu, qv, kb, vt, pb, mask, xb);
    // out = X*Wo^T + bo (f32)
    proj_gemm<3><<<dim3(512, 8), blk, 0, stream>>>(xb, wo, bo, nullptr, nullptr, d_out, nullptr, 8192);
}
